// LMU_14224931684544
// MI455X (gfx1250) — matrix operand order repaired and checked by reading, not compiled, not run on hardware
//
#include <hip/hip_runtime.h>
#include <math.h>

// ---------------- problem dimensions (from reference) ----------------
#define T_STEPS 1000
#define B_SZ    256
#define D_IN    128
#define H_DIM   512
#define M_DIM   256
#define O_DIM   128
#define K_ALL   (D_IN + H_DIM + M_DIM)   // 896

// ---------------- WMMA vector types (probe-verified) -----------------
typedef __attribute__((ext_vector_type(16))) __bf16        v16bf;
typedef __attribute__((ext_vector_type(8)))  float         v8f;
typedef __attribute__((ext_vector_type(4)))  float         v4f;
typedef __attribute__((ext_vector_type(16))) unsigned short v16u;
typedef __attribute__((ext_vector_type(8)))  unsigned short v8u;

// global (addrspace(1)) element/vector types: keep weight traffic on
// global_load_b128 (LOADcnt only), decoupled from LDS DScnt stream.
typedef __attribute__((address_space(1))) const unsigned short gus_t;
typedef __attribute__((address_space(1))) const v16u           gv16u_t;

__device__ __forceinline__ unsigned short f2bf(float f) {
    union { float f; unsigned u; } v; v.f = f;
    unsigned r = 0x7fffu + ((v.u >> 16) & 1u);      // round-to-nearest-even
    return (unsigned short)((v.u + r) >> 16);
}
__device__ __forceinline__ float bf2f(unsigned short h) {
    union { unsigned u; float f; } v; v.u = ((unsigned)h) << 16;
    return v.f;
}

// ---------------- workspace layout (bf16 weights) ---------------------
#define WS_WCAT   0          // [H_DIM][K_ALL] bf16 : 917504 B
#define WS_AB     917504     // [M_DIM][M_DIM] bf16 : 131072 B
#define WS_WOUTB  1048576    // [O_DIM][H_DIM] bf16 : 131072 B
#define WS_EVEC   1179648    // [K_ALL] f32          :   3584 B
#define WS_BVEC   1183232    // [M_DIM] f32          :   1024 B

// ======================================================================
// prep: f32 -> bf16 weight packing (runs every call; deterministic)
// ======================================================================
__global__ void lmu_prep(const float* __restrict__ Wx, const float* __restrict__ Wh,
                         const float* __restrict__ Wm, const float* __restrict__ Wout,
                         const float* __restrict__ A,  const float* __restrict__ ex,
                         const float* __restrict__ eh, const float* __restrict__ em,
                         const float* __restrict__ Bv,
                         unsigned short* __restrict__ Wcat, unsigned short* __restrict__ Ab,
                         unsigned short* __restrict__ Woutb, float* __restrict__ evec,
                         float* __restrict__ Bvec)
{
    int idx = blockIdx.x * blockDim.x + threadIdx.x;
    int stride = gridDim.x * blockDim.x;
    for (int i = idx; i < H_DIM * K_ALL; i += stride) {
        int n = i / K_ALL, k = i % K_ALL;
        float v = (k < D_IN)          ? Wx[n * D_IN + k]
                : (k < D_IN + H_DIM)  ? Wh[n * H_DIM + (k - D_IN)]
                                      : Wm[n * M_DIM + (k - D_IN - H_DIM)];
        Wcat[i] = f2bf(v);
    }
    for (int i = idx; i < M_DIM * M_DIM; i += stride) Ab[i]    = f2bf(A[i]);
    for (int i = idx; i < O_DIM * H_DIM; i += stride) Woutb[i] = f2bf(Wout[i]);
    for (int i = idx; i < K_ALL; i += stride)
        evec[i] = (i < D_IN) ? ex[i] : (i < D_IN + H_DIM) ? eh[i - D_IN]
                                                          : em[i - D_IN - H_DIM];
    for (int i = idx; i < M_DIM; i += stride) Bvec[i] = Bv[i];
}

// ======================================================================
// main persistent scan kernel: 16 blocks x 512 threads (16 waves)
// block b owns batch rows [16b, 16b+16): fully independent recurrence.
// ======================================================================
#define ZSTRIDE 1672                 // halfwords per LDS row (836 dw % 64 = 4 -> no conflicts)
#define XOFF    0
#define HOFF(buf) ((buf) ? 640 : 128)
#define MOFF(buf) ((buf) ? 1408 : 1152)

// A-fragment (16x32 bf16, contiguous k order, same as B):
// lane<16 holds K = +0..15 of row (lane&15); lane>=16 holds K = +16..31.
__device__ __forceinline__ v16bf load_afrag(const unsigned short* zsect, int h8) {
    v8u lo = *(const v8u*)(zsect + 2 * h8);
    v8u hi = *(const v8u*)(zsect + 2 * h8 + 8);
    v16u a;
#pragma unroll
    for (int i = 0; i < 8; ++i) { a[i] = lo[i]; a[i + 8] = hi[i]; }
    return __builtin_bit_cast(v16bf, a);
}

// B-fragment from global weights: lane holds 16 contiguous K values of
// weight row (= output col); Bp pre-offset per lane.
__device__ __forceinline__ v16bf load_bfrag(const gus_t* Bp, int k0) {
    return __builtin_bit_cast(v16bf, *(gv16u_t*)(Bp + k0));
}

__device__ __forceinline__ v8f wmma_bf16(v16bf a, v16bf b, v8f c) {
    return __builtin_amdgcn_wmma_f32_16x16x32_bf16(false, a, false, b, (short)0, c,
                                                   false, false);
}

// h-GEMM K-step s (0..27) -> LDS section offset + weight column offset
__device__ __forceinline__ void hstep_map(int s, int hoff_r, int moff_w,
                                          int& zoff, int& woff) {
    if (s < 4)       { zoff = XOFF   + 32 * s;        woff = 32 * s; }
    else if (s < 20) { zoff = hoff_r + 32 * (s - 4);  woff = D_IN + 32 * (s - 4); }
    else             { zoff = moff_w + 32 * (s - 20); woff = D_IN + H_DIM + 32 * (s - 20); }
}

__global__ __launch_bounds__(512, 1)
void lmu_scan(const float* __restrict__ x,
              const unsigned short* __restrict__ Wcat_g,
              const unsigned short* __restrict__ Ab_g,
              const unsigned short* __restrict__ Woutb_g,
              const float* __restrict__ evec,
              const float* __restrict__ Bvec,
              float* __restrict__ out)
{
    __shared__ unsigned short z[16 * ZSTRIDE];   // [x | h0 | h1 | m0 | m1] bf16 per row
    __shared__ float u_s[16];

    const int tid  = threadIdx.x;
    const int lane = tid & 31;
    const int wave = tid >> 5;              // 0..15
    const int r0   = blockIdx.x * 16;       // owned batch rows

    for (int i = tid; i < 16 * ZSTRIDE; i += 512) z[i] = 0;   // h0 = m0 = 0
    __syncthreads();

    const int mrow  = lane & 15;            // A row / D column within tile
    const int khalf = lane >> 4;            // 0 or 1
    const int h8    = khalf << 3;
    const int bofs  = khalf << 4;           // B-fragment K-half
    const unsigned short* zArow = &z[mrow * ZSTRIDE];

    for (int t = 0; t < T_STEPS; ++t) {
        const int p = t & 1;
        const int hoff_r = HOFF(p),  hoff_w = HOFF(p ^ 1);
        const int moff_r = MOFF(p),  moff_w = MOFF(p ^ 1);

        // Launder weight bases each step (prevents cross-iteration hoist +
        // scratch spill) while keeping them addrspace(1) => global_load_b128.
        gus_t* Wcat  = (gus_t*)(unsigned long long)Wcat_g;
        gus_t* Ab    = (gus_t*)(unsigned long long)Ab_g;
        gus_t* Woutb = (gus_t*)(unsigned long long)Woutb_g;
        asm volatile("" : "+s"(Wcat), "+s"(Ab), "+s"(Woutb));

        // ---- stage x_t tile: 16x128 f32 -> bf16 LDS (1 wave per row) ----
        {
            const int row = wave;
            const int c4  = lane * 4;
            const v4f v = __builtin_nontemporal_load(
                (const v4f*)(x + ((size_t)t * B_SZ + r0 + row) * D_IN + c4));
            unsigned short* zp = &z[row * ZSTRIDE + XOFF + c4];
            zp[0] = f2bf(v.x); zp[1] = f2bf(v.y); zp[2] = f2bf(v.z); zp[3] = f2bf(v.w);
            if (t + 1 < T_STEPS)   // warm L2 for next step (global_prefetch_b8)
                __builtin_prefetch(x + ((size_t)(t + 1) * B_SZ + r0 + row) * D_IN + c4, 0, 1);
        }
        __syncthreads();

        // ---- u[row] = [x|h|m]_row . evec  (one wave per row, shuffle reduce) ----
        {
            const unsigned short* zr = &z[wave * ZSTRIDE];
            float s = 0.f;
#pragma unroll
            for (int k = 0; k < D_IN; k += 32)
                s += bf2f(zr[XOFF + k + lane]) * evec[k + lane];
#pragma unroll
            for (int k = 0; k < H_DIM; k += 32)
                s += bf2f(zr[hoff_r + k + lane]) * evec[D_IN + k + lane];
#pragma unroll
            for (int k = 0; k < M_DIM; k += 32)
                s += bf2f(zr[moff_r + k + lane]) * evec[D_IN + H_DIM + k + lane];
#pragma unroll
            for (int off = 16; off > 0; off >>= 1) s += __shfl_xor(s, off, 32);
            if (lane == 0) u_s[wave] = s;
        }
        __syncthreads();

        // ---- m_new = m_old @ A^T + u * Bvec : one 16x16 tile per wave ----
        {
            const int n0 = wave * 16;
            gus_t* Bp = Ab + (size_t)(n0 + mrow) * M_DIM + bofs;
            v8f acc = {};
            v16bf a  = load_afrag(zArow + moff_r, h8);
            v16bf b  = load_bfrag(Bp, 0);
#pragma unroll
            for (int s = 0; s < 8; ++s) {            // K = 256
                v16bf an = a, bn = b;
                if (s + 1 < 8) {
                    an = load_afrag(zArow + moff_r + 32 * (s + 1), h8);
                    bn = load_bfrag(Bp, 32 * (s + 1));
                }
                acc = wmma_bf16(a, b, acc);
                a = an; b = bn;
            }
            const float bn = Bvec[n0 + mrow];
#pragma unroll
            for (int v = 0; v < 8; ++v) {
                const int Mr = h8 + v;
                z[Mr * ZSTRIDE + moff_w + n0 + mrow] = f2bf(acc[v] + u_s[Mr] * bn);
            }
        }
        __syncthreads();

        // ---- h_new = tanh([x | h_old | m_new] @ Wcat^T) : 2 tiles per wave ----
        {
            const int n0a = wave * 32, n0b = n0a + 16;
            gus_t* Bpa = Wcat + (size_t)(n0a + mrow) * K_ALL + bofs;
            gus_t* Bpb = Wcat + (size_t)(n0b + mrow) * K_ALL + bofs;
            v8f acc0 = {}, acc1 = {};
            int zoff, woff;
            hstep_map(0, hoff_r, moff_w, zoff, woff);
            v16bf a  = load_afrag(zArow + zoff, h8);
            v16bf b0 = load_bfrag(Bpa, woff);
            v16bf b1 = load_bfrag(Bpb, woff);
#pragma unroll
            for (int s = 0; s < 28; ++s) {           // K = 896
                v16bf an = a, bn0 = b0, bn1 = b1;
                if (s + 1 < 28) {
                    hstep_map(s + 1, hoff_r, moff_w, zoff, woff);
                    an  = load_afrag(zArow + zoff, h8);
                    bn0 = load_bfrag(Bpa, woff);
                    bn1 = load_bfrag(Bpb, woff);
                }
                acc0 = wmma_bf16(a, b0, acc0);
                acc1 = wmma_bf16(a, b1, acc1);
                a = an; b0 = bn0; b1 = bn1;
            }
#pragma unroll
            for (int v = 0; v < 8; ++v) {
                const int Mr = h8 + v;
                z[Mr * ZSTRIDE + hoff_w + n0a + mrow] = f2bf(tanhf(acc0[v]));
                z[Mr * ZSTRIDE + hoff_w + n0b + mrow] = f2bf(tanhf(acc1[v]));
            }
        }
        __syncthreads();

        // ---- out_t = h_new @ Wout^T : 8 tiles, waves 0..7 ----
        if (wave < 8) {
            const int n0 = wave * 16;
            gus_t* Bp = Woutb + (size_t)(n0 + mrow) * H_DIM + bofs;
            v8f acc = {};
            v16bf a = load_afrag(zArow + hoff_w, h8);
            v16bf b = load_bfrag(Bp, 0);
#pragma unroll
            for (int s = 0; s < 16; ++s) {           // K = 512
                v16bf an = a, bn = b;
                if (s + 1 < 16) {
                    an = load_afrag(zArow + hoff_w + 32 * (s + 1), h8);
                    bn = load_bfrag(Bp, 32 * (s + 1));
                }
                acc = wmma_bf16(a, b, acc);
                a = an; b = bn;
            }
            float* op = out + (size_t)t * B_SZ * O_DIM;
#pragma unroll
            for (int v = 0; v < 8; ++v) {
                const int Mr = h8 + v;
                __builtin_nontemporal_store(acc[v],
                    op + (size_t)(r0 + Mr) * O_DIM + n0 + mrow);
            }
        }
        __syncthreads();
    }
}

// ======================================================================
extern "C" void kernel_launch(void* const* d_in, const int* in_sizes, int n_in,
                              void* d_out, int out_size, void* d_ws, size_t ws_size,
                              hipStream_t stream)
{
    (void)in_sizes; (void)n_in; (void)out_size; (void)ws_size;
    const float* x    = (const float*)d_in[0];
    const float* e_x  = (const float*)d_in[1];
    const float* e_h  = (const float*)d_in[2];
    const float* e_m  = (const float*)d_in[3];
    const float* Wx   = (const float*)d_in[4];
    const float* Wh   = (const float*)d_in[5];
    const float* Wm   = (const float*)d_in[6];
    const float* Wout = (const float*)d_in[7];
    const float* A    = (const float*)d_in[8];
    const float* Bv   = (const float*)d_in[9];

    char* ws = (char*)d_ws;
    unsigned short* Wcat  = (unsigned short*)(ws + WS_WCAT);
    unsigned short* Ab    = (unsigned short*)(ws + WS_AB);
    unsigned short* Woutb = (unsigned short*)(ws + WS_WOUTB);
    float* evec = (float*)(ws + WS_EVEC);
    float* Bvec = (float*)(ws + WS_BVEC);

    lmu_prep<<<512, 256, 0, stream>>>(Wx, Wh, Wm, Wout, A, e_x, e_h, e_m, Bv,
                                      Wcat, Ab, Woutb, evec, Bvec);
    lmu_scan<<<16, 512, 0, stream>>>(x, Wcat, Ab, Woutb, evec, Bvec, (float*)d_out);
}
